// BigBirdAttention_56796647522735
// MI455X (gfx1250) — matrix operand order repaired and checked by reading, not compiled, not run on hardware
//
#include <hip/hip_runtime.h>

typedef __attribute__((ext_vector_type(16))) __bf16          v16bf;
typedef __attribute__((ext_vector_type(2)))  __bf16          v2bf;
typedef __attribute__((ext_vector_type(16))) unsigned short  v16us;
typedef __attribute__((ext_vector_type(8)))  float           v8f;
typedef __attribute__((ext_vector_type(2)))  float           v2f;
typedef __attribute__((ext_vector_type(8)))  unsigned int    v8u;

#define BB 4
#define TT 4096
#define HH 16
#define DD 64
#define GG 64
#define BS 64
#define NB 63            // (TT-GG)/BS
#define CK 64            // keys per chunk
#define LDK 80           // K LDS row stride (bf16 elems): 160 B, 32-aligned
#define LDV 80           // vT LDS row stride (bf16 elems): 160 B, 32-aligned
#define LDP 80           // P LDS row stride (bf16 elems): 160 B, 32-aligned

// packed f32x2 -> bf16x2 through IR fptrunc: lets the backend pick v_cvt_pk_bf16_f32
__device__ __forceinline__ unsigned cvtpk(float lo, float hi) {
  v2f t = { lo, hi };
  v2bf r = __builtin_convertvector(t, v2bf);
  return __builtin_bit_cast(unsigned, r);
}
__device__ __forceinline__ v16bf ld16bf(const unsigned short* p) {
  return __builtin_bit_cast(v16bf, *(const v16us*)p);   // 32-byte aligned vector LDS load
}

// opaque max: avoids the compiler's canonicalize (v_max x,x) companions
__device__ __forceinline__ float vmax(float a, float b) {
  float d;
  asm("v_max_num_f32 %0, %1, %2" : "=v"(d) : "v"(a), "v"(b));
  return d;
}

// hardware exp2 (v_exp_f32)
__device__ __forceinline__ float hexp2(float x) {
  return __builtin_amdgcn_exp2f(x);
}

// ---- 16-lane-group reductions in pure VALU via DPP16 ----
template <int CTRL>
__device__ __forceinline__ float dpp_mov(float x) {
  return __builtin_bit_cast(float,
      __builtin_amdgcn_update_dpp(0, __builtin_bit_cast(int, x), CTRL, 0xF, 0xF, true));
}
__device__ __forceinline__ float red_max16(float x) {
  x = vmax(x, dpp_mov<0xB1>(x));     // quad_perm(1,0,3,2)  : xor-1
  x = vmax(x, dpp_mov<0x4E>(x));     // quad_perm(2,3,0,1)  : xor-2
  x = vmax(x, dpp_mov<0x141>(x));    // row_half_mirror
  x = vmax(x, dpp_mov<0x140>(x));    // row_mirror
  return x;
}
__device__ __forceinline__ float red_sum16(float x) {
  x += dpp_mov<0xB1>(x);
  x += dpp_mov<0x4E>(x);
  x += dpp_mov<0x141>(x);
  x += dpp_mov<0x140>(x);
  return x;
}

__launch_bounds__(128)
__global__ void bigbird_fused_wmma(const float* __restrict__ qg,
                                   const float* __restrict__ kg,
                                   const float* __restrict__ vg,
                                   const int*   __restrict__ rnd,
                                   float* __restrict__ outg) {
  __shared__ unsigned short klds[2][CK][LDK];    // K chunk (double-buffered), [key][d]
  __shared__ unsigned short vT  [2][DD][LDV];    // V chunk (double-buffered), [d][key]
  __shared__ unsigned short plds[4][16][LDP];    // per-wave P re-layout buffer (16x64)

  const int tid  = threadIdx.x;
  const int wave = tid >> 5;
  const int lane = tid & 31;
  const int half = lane >> 4;
  const int l16  = lane & 15;

  const int u  = blockIdx.x;
  const int b  = u / (HH * (NB + 1));
  const int r0 = u % (HH * (NB + 1));
  const int h  = r0 / (NB + 1);
  const int nn = r0 % (NB + 1);
  const bool gmode = (nn == 0);              // global-rows mode: 64 q rows vs all 4096 keys
  const int qt0 = gmode ? 0 : (GG + (nn - 1) * BS);
  const int nchunks = gmode ? (TT / CK) : 5;   // 5*64 = 64 local + 64 global + 192 random

  // fold softmax scale AND log2(e) into the Q pre-scale: scores live in exp2 domain
  const float qscale = 0.125f * 1.4426950408889634f;

  // staging thread mapping: 128 threads cover 64 keys x 64 d
  const int kloc  = tid >> 1;         // 0..63 key slot
  const int dpart = (tid & 1) * 32;   // 0/32

  // ---- Q slice (16 rows per wave) into WMMA A-layout registers, pre-scaled ----
  v16bf qa0, qa1;
  {
    const int row = wave * 16 + l16;
    const float* qp = qg + (((long)b * TT + (qt0 + row)) * HH + h) * DD;
    float4 f0 = *(const float4*)(qp + 16 * half);
    float4 f1 = *(const float4*)(qp + 16 * half + 4);
    float4 f2 = *(const float4*)(qp + 16 * half + 8);
    float4 f3 = *(const float4*)(qp + 16 * half + 12);
    float4 f4 = *(const float4*)(qp + 32 + 16 * half);
    float4 f5 = *(const float4*)(qp + 32 + 16 * half + 4);
    float4 f6 = *(const float4*)(qp + 32 + 16 * half + 8);
    float4 f7 = *(const float4*)(qp + 32 + 16 * half + 12);
    const float s = qscale;
    v8u w0, w1;
    w0[0] = cvtpk(f0.x*s, f0.y*s); w0[1] = cvtpk(f0.z*s, f0.w*s);
    w0[2] = cvtpk(f1.x*s, f1.y*s); w0[3] = cvtpk(f1.z*s, f1.w*s);
    w0[4] = cvtpk(f2.x*s, f2.y*s); w0[5] = cvtpk(f2.z*s, f2.w*s);
    w0[6] = cvtpk(f3.x*s, f3.y*s); w0[7] = cvtpk(f3.z*s, f3.w*s);
    w1[0] = cvtpk(f4.x*s, f4.y*s); w1[1] = cvtpk(f4.z*s, f4.w*s);
    w1[2] = cvtpk(f5.x*s, f5.y*s); w1[3] = cvtpk(f5.z*s, f5.w*s);
    w1[4] = cvtpk(f6.x*s, f6.y*s); w1[5] = cvtpk(f6.z*s, f6.w*s);
    w1[6] = cvtpk(f7.x*s, f7.y*s); w1[7] = cvtpk(f7.z*s, f7.w*s);
    qa0 = __builtin_bit_cast(v16bf, w0);
    qa1 = __builtin_bit_cast(v16bf, w1);
  }

  float m[8], l[8];
  v8f o0 = {0,0,0,0,0,0,0,0}, o1 = {0,0,0,0,0,0,0,0};
  v8f o2 = {0,0,0,0,0,0,0,0}, o3 = {0,0,0,0,0,0,0,0};
#pragma unroll
  for (int j = 0; j < 8; ++j) { m[j] = -3.0e38f; l[j] = 0.0f; }

  // ---- pipelined global->LDS staging (64 keys per chunk) ----
  float4 kr[8], vr[8];
  auto load_regs = [&](int c) {
    const int idx = c * CK + kloc;
    int kt;
    if (gmode)          kt = idx;               // sequential keys
    else if (idx < 64)  kt = qt0 + idx;         // local block keys
    else if (idx < 128) kt = idx - 64;          // global keys 0..63
    else                kt = rnd[idx - 128];    // random keys
    const long kb = (((long)b * TT + kt) * HH + h) * DD + dpart;
    const float4* kp = (const float4*)(kg + kb);
    const float4* vp = (const float4*)(vg + kb);
#pragma unroll
    for (int i = 0; i < 8; ++i) { kr[i] = kp[i]; vr[i] = vp[i]; }
  };
  auto store_lds = [&](int buf) {
    // K: row-major, 32 contiguous bf16 -> four uint4 stores (32-byte aligned)
#pragma unroll
    for (int i = 0; i < 4; ++i) {
      uint4 w;
      w.x = cvtpk(kr[2*i].x,   kr[2*i].y);   w.y = cvtpk(kr[2*i].z,   kr[2*i].w);
      w.z = cvtpk(kr[2*i+1].x, kr[2*i+1].y); w.w = cvtpk(kr[2*i+1].z, kr[2*i+1].w);
      *(uint4*)&klds[buf][kloc][dpart + 8*i] = w;
    }
    // V: transposed scatter (d-major); one packed cvt per pair, hi via >>16
    float vv[32] = { vr[0].x,vr[0].y,vr[0].z,vr[0].w, vr[1].x,vr[1].y,vr[1].z,vr[1].w,
                     vr[2].x,vr[2].y,vr[2].z,vr[2].w, vr[3].x,vr[3].y,vr[3].z,vr[3].w,
                     vr[4].x,vr[4].y,vr[4].z,vr[4].w, vr[5].x,vr[5].y,vr[5].z,vr[5].w,
                     vr[6].x,vr[6].y,vr[6].z,vr[6].w, vr[7].x,vr[7].y,vr[7].z,vr[7].w };
#pragma unroll
    for (int i = 0; i < 16; ++i) {
      const unsigned w = cvtpk(vv[2*i], vv[2*i+1]);
      vT[buf][dpart + 2*i][kloc]     = (unsigned short)w;
      vT[buf][dpart + 2*i + 1][kloc] = (unsigned short)(w >> 16);  // -> ds_store_b16_d16_hi
    }
  };

  // prologue: stage chunk 0
  load_regs(0);
  store_lds(0);
  __syncthreads();

  for (int c = 0; c < nchunks; ++c) {
    const int buf = c & 1;
    const bool more = (c + 1 < nchunks);
    if (more) load_regs(c + 1);      // global loads in flight while we compute

    // ---- S = (Q*qscale) K^T : 16 rows x 64 keys; grouped by shared A operand ----
    v8f s0 = {0,0,0,0,0,0,0,0}, s1 = {0,0,0,0,0,0,0,0};
    v8f s2 = {0,0,0,0,0,0,0,0}, s3 = {0,0,0,0,0,0,0,0};
    {
      v16bf bk;
      bk = ld16bf(&klds[buf][l16][16 * half]);
      s0 = __builtin_amdgcn_wmma_f32_16x16x32_bf16(false, qa0, false, bk, (short)0, s0, false, false);
      bk = ld16bf(&klds[buf][16 + l16][16 * half]);
      s1 = __builtin_amdgcn_wmma_f32_16x16x32_bf16(false, qa0, false, bk, (short)0, s1, false, false);
      bk = ld16bf(&klds[buf][32 + l16][16 * half]);
      s2 = __builtin_amdgcn_wmma_f32_16x16x32_bf16(false, qa0, false, bk, (short)0, s2, false, false);
      bk = ld16bf(&klds[buf][48 + l16][16 * half]);
      s3 = __builtin_amdgcn_wmma_f32_16x16x32_bf16(false, qa0, false, bk, (short)0, s3, false, false);
      bk = ld16bf(&klds[buf][l16][32 + 16 * half]);
      s0 = __builtin_amdgcn_wmma_f32_16x16x32_bf16(false, qa1, false, bk, (short)0, s0, false, false);
      bk = ld16bf(&klds[buf][16 + l16][32 + 16 * half]);
      s1 = __builtin_amdgcn_wmma_f32_16x16x32_bf16(false, qa1, false, bk, (short)0, s1, false, false);
      bk = ld16bf(&klds[buf][32 + l16][32 + 16 * half]);
      s2 = __builtin_amdgcn_wmma_f32_16x16x32_bf16(false, qa1, false, bk, (short)0, s2, false, false);
      bk = ld16bf(&klds[buf][48 + l16][32 + 16 * half]);
      s3 = __builtin_amdgcn_wmma_f32_16x16x32_bf16(false, qa1, false, bk, (short)0, s3, false, false);
    }

    // ---- online softmax in exp2 domain (DPP16 butterflies, no LDS) ----
    const int poff = l16;               // key-indexed store idx -> contiguous A-layout read
#pragma unroll
    for (int j = 0; j < 8; ++j) {
      const float sv0 = s0[j], sv1 = s1[j], sv2 = s2[j], sv3 = s3[j];
      const float mx = red_max16(vmax(vmax(sv0, sv1), vmax(sv2, sv3)));
      const float mn = vmax(m[j], mx);
      const float sc = hexp2(m[j] - mn);
      const float p0 = hexp2(sv0 - mn);
      const float p1 = hexp2(sv1 - mn);
      const float p2 = hexp2(sv2 - mn);
      const float p3 = hexp2(sv3 - mn);
      const float ps = red_sum16((p0 + p1) + (p2 + p3));
      l[j] = l[j] * sc + ps;
      m[j] = mn;
      o0[j] *= sc; o1[j] *= sc; o2[j] *= sc; o3[j] *= sc;
      const int prow = j + 8 * half;           // C-layout row for this element
      const unsigned wlo = cvtpk(p0, p1);      // keys c / 16+c
      const unsigned whi = cvtpk(p2, p3);      // keys 32+c / 48+c
      plds[wave][prow][poff]          = (unsigned short)wlo;
      plds[wave][prow][poff + 16]     = (unsigned short)(wlo >> 16);
      plds[wave][prow][32 + poff]     = (unsigned short)whi;
      plds[wave][prow][32 + poff + 16] = (unsigned short)(whi >> 16);
    }

    // ---- O += P V (two P A-matrices covering keys 0..31 / 32..63) ----
    {
      const v16bf pa0 = ld16bf(&plds[wave][l16][16 * half]);
      const v16bf pa1 = ld16bf(&plds[wave][l16][32 + 16 * half]);
      v16bf bv;
      bv = ld16bf(&vT[buf][l16][16 * half]);
      o0 = __builtin_amdgcn_wmma_f32_16x16x32_bf16(false, pa0, false, bv, (short)0, o0, false, false);
      bv = ld16bf(&vT[buf][16 + l16][16 * half]);
      o1 = __builtin_amdgcn_wmma_f32_16x16x32_bf16(false, pa0, false, bv, (short)0, o1, false, false);
      bv = ld16bf(&vT[buf][32 + l16][16 * half]);
      o2 = __builtin_amdgcn_wmma_f32_16x16x32_bf16(false, pa0, false, bv, (short)0, o2, false, false);
      bv = ld16bf(&vT[buf][48 + l16][16 * half]);
      o3 = __builtin_amdgcn_wmma_f32_16x16x32_bf16(false, pa0, false, bv, (short)0, o3, false, false);
      bv = ld16bf(&vT[buf][l16][32 + 16 * half]);
      o0 = __builtin_amdgcn_wmma_f32_16x16x32_bf16(false, pa1, false, bv, (short)0, o0, false, false);
      bv = ld16bf(&vT[buf][16 + l16][32 + 16 * half]);
      o1 = __builtin_amdgcn_wmma_f32_16x16x32_bf16(false, pa1, false, bv, (short)0, o1, false, false);
      bv = ld16bf(&vT[buf][32 + l16][32 + 16 * half]);
      o2 = __builtin_amdgcn_wmma_f32_16x16x32_bf16(false, pa1, false, bv, (short)0, o2, false, false);
      bv = ld16bf(&vT[buf][48 + l16][32 + 16 * half]);
      o3 = __builtin_amdgcn_wmma_f32_16x16x32_bf16(false, pa1, false, bv, (short)0, o3, false, false);
    }

    if (more) store_lds(buf ^ 1);    // fill the other buffer for next iteration
    __syncthreads();
  }

  // ---- normalize and write out (C layout -> [b,t,h,d]) ----
#pragma unroll
  for (int j = 0; j < 8; ++j) {
    const int row = wave * 16 + j + 8 * half;
    const long ob = (((long)b * TT + (qt0 + row)) * HH + h) * DD + l16;
    const float inv_l = 1.0f / l[j];
    outg[ob +  0] = o0[j] * inv_l;
    outg[ob + 16] = o1[j] * inv_l;
    outg[ob + 32] = o2[j] * inv_l;
    outg[ob + 48] = o3[j] * inv_l;
  }
}

extern "C" void kernel_launch(void* const* d_in, const int* in_sizes, int n_in,
                              void* d_out, int out_size, void* d_ws, size_t ws_size,
                              hipStream_t stream) {
  (void)in_sizes; (void)n_in; (void)out_size; (void)d_ws; (void)ws_size;
  const float* q   = (const float*)d_in[0];
  const float* k   = (const float*)d_in[1];
  const float* v   = (const float*)d_in[2];
  const int*   rnd = (const int*)d_in[3];
  float* out = (float*)d_out;
  dim3 grid(BB * HH * (NB + 1));   // 4 * 16 * 64 = 4096 workgroups
  bigbird_fused_wmma<<<grid, 128, 0, stream>>>(q, k, v, rnd, out);
}
